// Speller_32349693673665
// MI455X (gfx1250) — matrix operand order repaired and checked by reading, not compiled, not run on hardware
//
#include <hip/hip_runtime.h>
#include <stdint.h>
#include <math.h>

// ---------------- sizes ----------------
#define BB   128      // batch
#define TT   400      // listener timesteps
#define HH   512      // hidden
#define MLPD 256      // mlp dim
#define LL   150      // output steps
#define NL   32       // num labels
#define KX   1056     // 32 (tok) + 512 (context) + 512 (h)
#define G4   2048     // 4*H gate width

#define NWG   64
#define NTHR  256
#define NWAVES (NWG * (NTHR / 32))   // 512 waves

typedef __attribute__((ext_vector_type(16))) __bf16    v16bf;
typedef __attribute__((ext_vector_type(8)))  float     v8f;
typedef __attribute__((ext_vector_type(8)))  unsigned  v8u;

// ---------------- helpers ----------------
__device__ __forceinline__ unsigned short f2bf(float f) {
    union { float f; unsigned u; } x; x.f = f;
    unsigned u = x.u + 0x7fffu + ((x.u >> 16) & 1u);
    return (unsigned short)(u >> 16);
}
__device__ __forceinline__ float bf2f(unsigned short h) {
    union { unsigned u; float f; } x; x.u = ((unsigned)h) << 16;
    return x.f;
}
__device__ __forceinline__ float sigm(float x) { return 1.0f / (1.0f + __expf(-x)); }

// generic shared pointer -> LDS byte offset (low 32 bits of flat LDS address)
__device__ __forceinline__ unsigned lds_offset(const void* p) {
    return (unsigned)(uintptr_t)p;
}

// CDNA5 async copy: 16B per lane, global -> LDS, tracked by ASYNCcnt
__device__ __forceinline__ void async_ld_b128(unsigned lds_addr, unsigned long long gaddr) {
    asm volatile("global_load_async_to_lds_b128 %0, %1, off"
                 :: "v"(lds_addr), "v"(gaddr) : "memory");
}

// A fragment (16x32 bf16, rows m0..m0+15, cols k0..k0+31 of row-major [*,ld]):
// lanes 0-15: M=lane, VGPR0-3 = K 0..7, VGPR4-7 = K 8..15
// lanes 16-31: M=lane-16, VGPR0-3 = K 16..23, VGPR4-7 = K 24..31
__device__ __forceinline__ v16bf load_A(const unsigned short* base, int m0, int k0, int ld) {
    int lane = threadIdx.x & 31;
    int half = lane >> 4;
    const uint4* p = (const uint4*)(base + (size_t)(m0 + (lane & 15)) * ld + k0);
    uint4 lo = p[2 * half];
    uint4 hi = p[2 * half + 1];
    v8u u;
    u[0] = lo.x; u[1] = lo.y; u[2] = lo.z; u[3] = lo.w;
    u[4] = hi.x; u[5] = hi.y; u[6] = hi.z; u[7] = hi.w;
    return __builtin_bit_cast(v16bf, u);
}

// B fragment (32x16 bf16) for C = A * W^T with W row-major [N][K]:
// column n = W row n; lanes 0-15 hold K 0..15 (cols n0+lane), lanes 16-31 hold K 16..31
__device__ __forceinline__ v16bf load_B(const unsigned short* base, int n0, int k0, int ld) {
    int lane = threadIdx.x & 31;
    int half = lane >> 4;
    const uint4* p = (const uint4*)(base + (size_t)(n0 + (lane & 15)) * ld + k0 + half * 16);
    uint4 lo = p[0];
    uint4 hi = p[1];
    v8u u;
    u[0] = lo.x; u[1] = lo.y; u[2] = lo.z; u[3] = lo.w;
    u[4] = hi.x; u[5] = hi.y; u[6] = hi.z; u[7] = hi.w;
    return __builtin_bit_cast(v16bf, u);
}

// B fragment from LDS staging buffer: lane l holds its 16 bf16 at [l*16 .. l*16+15]
__device__ __forceinline__ v16bf load_B_sm(const unsigned short* sm) {
    int lane = threadIdx.x & 31;
    const uint4* p = (const uint4*)(sm + lane * 16);
    uint4 lo = p[0];
    uint4 hi = p[1];
    v8u u;
    u[0] = lo.x; u[1] = lo.y; u[2] = lo.z; u[3] = lo.w;
    u[4] = hi.x; u[5] = hi.y; u[6] = hi.z; u[7] = hi.w;
    return __builtin_bit_cast(v16bf, u);
}

// C/D layout: VGPR v, lanes 0-15 -> M = m0+v, lanes 16-31 -> M = m0+8+v; N = n0 + (lane&15)
__device__ __forceinline__ void store_C(float* C, int m0, int n0, int ldc, v8f acc) {
    int lane = threadIdx.x & 31;
    int half = lane >> 4;
    int col  = n0 + (lane & 15);
#pragma unroll
    for (int v = 0; v < 8; ++v)
        C[(size_t)(m0 + half * 8 + v) * ldc + col] = acc[v];
}

__device__ __forceinline__ v8f wmma_bf16(v16bf a, v16bf b, v8f c) {
    return __builtin_amdgcn_wmma_f32_16x16x32_bf16(false, a, false, b, (short)0, c, false, false);
}

// global software barrier (persistent-kernel phase sync)
__device__ __forceinline__ void gbar(int* bar, int& target) {
    __syncthreads();
    if (threadIdx.x == 0) {
        __threadfence();
        atomicAdd(bar, 1);
        target += NWG;
        while (__hip_atomic_load(bar, __ATOMIC_ACQUIRE, __HIP_MEMORY_SCOPE_AGENT) < target)
            __builtin_amdgcn_s_sleep(2);
        __threadfence();
    }
    __syncthreads();
}

// ---------------- kernel 1: weight prep (bf16 conversion / repack) ----------------
__global__ void __launch_bounds__(256) prep_kernel(
    const float* __restrict__ W_ih, const float* __restrict__ W_hh,
    const float* __restrict__ b_ih, const float* __restrict__ b_hh,
    const float* __restrict__ phi_W, const float* __restrict__ psi_W,
    const float* __restrict__ fc_W, const float* __restrict__ lf,
    unsigned short* __restrict__ Wcat, unsigned short* __restrict__ phiWb,
    unsigned short* __restrict__ psiWb, unsigned short* __restrict__ fcWb,
    float* __restrict__ bias, unsigned short* __restrict__ lfb)
{
    int stride = gridDim.x * blockDim.x;
    int t0 = blockIdx.x * blockDim.x + threadIdx.x;
    // Wcat[n][k]: k<544 -> W_ih (tok+context), k>=544 -> W_hh (h)
    for (int i = t0; i < G4 * KX; i += stride) {
        int n = i / KX, k = i - n * KX;
        float v = (k < 544) ? W_ih[n * 544 + k] : W_hh[n * 512 + (k - 544)];
        Wcat[i] = f2bf(v);
    }
    for (int i = t0; i < MLPD * HH; i += stride) phiWb[i] = f2bf(phi_W[i]);
    for (int i = t0; i < MLPD * HH; i += stride) psiWb[i] = f2bf(psi_W[i]);
    // fc input from xh is [context | h]; original fc_W cols are [h | context] -> permute
    for (int i = t0; i < NL * 1024; i += stride) {
        int n = i >> 10, k = i & 1023;
        int sk = (k < 512) ? (512 + k) : (k - 512);
        fcWb[i] = f2bf(fc_W[n * 1024 + sk]);
    }
    for (int i = t0; i < G4; i += stride) bias[i] = b_ih[i] + b_hh[i];
    for (size_t i = t0; i < (size_t)BB * TT * HH; i += stride) lfb[i] = f2bf(lf[i]);
}

// ---------------- kernel 2: comp_listener = relu(lf @ psi_W^T + psi_b), bf16 out ----------------
__global__ void __launch_bounds__(256) psi_gemm_kernel(
    const unsigned short* __restrict__ lfb, const unsigned short* __restrict__ psiW,
    const float* __restrict__ psi_b, unsigned short* __restrict__ cl)
{
    int gwave = blockIdx.x * (blockDim.x >> 5) + (threadIdx.x >> 5);
    int nwaves = gridDim.x * (blockDim.x >> 5);
    int lane = threadIdx.x & 31;
    int half = lane >> 4;
    const int MT = (BB * TT) / 16;     // 3200 row tiles
    const int NT = MLPD / 16;          // 16 col tiles
    for (int t = gwave; t < MT * NT; t += nwaves) {
        int mt = t / NT, nt = t - mt * NT;
        v8f acc = {0.f, 0.f, 0.f, 0.f, 0.f, 0.f, 0.f, 0.f};
#pragma unroll 4
        for (int k0 = 0; k0 < HH; k0 += 32) {
            v16bf a = load_A(lfb, mt * 16, k0, HH);
            v16bf b = load_B(psiW, nt * 16, k0, HH);
            acc = wmma_bf16(a, b, acc);
        }
        int col = nt * 16 + (lane & 15);
        float bv = psi_b[col];
#pragma unroll
        for (int v = 0; v < 8; ++v) {
            float r = acc[v] + bv;
            r = r > 0.f ? r : 0.f;
            cl[(size_t)(mt * 16 + half * 8 + v) * MLPD + col] = f2bf(r);
        }
    }
}

// ---------------- kernel 3: init state ----------------
__global__ void __launch_bounds__(256) init_kernel(
    const unsigned short* __restrict__ lfb, unsigned short* __restrict__ xh,
    float* __restrict__ cst)
{
    int stride = gridDim.x * blockDim.x;
    int t0 = blockIdx.x * blockDim.x + threadIdx.x;
    for (int i = t0; i < BB * KX; i += stride) {
        int b = i / KX, k = i - b * KX;
        unsigned short v;
        if (k < 32)       v = (k == 0) ? f2bf(1.0f) : (unsigned short)0;   // one-hot SOS=0
        else if (k < 544) v = lfb[(size_t)b * TT * HH + (k - 32)];         // listener_feats[b,0,:]
        else              v = 0;                                            // h = 0
        xh[i] = v;
    }
    for (int i = t0; i < BB * HH; i += stride) cst[i] = 0.f;
}

// ---------------- kernel 4: persistent decode over 150 steps ----------------
__global__ void __launch_bounds__(256, 1) decode_kernel(
    const unsigned short* __restrict__ lfb,   // [B*T*H] bf16
    const unsigned short* __restrict__ cl,    // [B*T*MLP] bf16
    const unsigned short* __restrict__ Wcat,  // [2048*1056] bf16
    const unsigned short* __restrict__ phiW,  // [256*512] bf16
    const unsigned short* __restrict__ fcW,   // [32*1024] bf16
    const float* __restrict__ bias,           // [2048]
    const float* __restrict__ phi_b,          // [256]
    const float* __restrict__ fc_b,           // [32]
    const int*   __restrict__ gt,             // [B*L]
    unsigned short* __restrict__ xh,          // [B*KX] bf16
    float* __restrict__ cst,                  // [B*H]
    float* __restrict__ gates,                // [B*2048]
    float* __restrict__ qbuf,                 // [B*256]
    float* __restrict__ logits,               // [B*32]
    float* __restrict__ out,                  // [B*L*32]
    int* bar)
{
    __shared__ float sm_q[MLPD];
    __shared__ float sm_e[TT];
    __shared__ float sm_red[NTHR];
    // double-buffered async staging for the gates-GEMM B operand:
    // [buffer][local n-tile][16 rows x 32 k] in per-lane fragment layout
    __shared__ __align__(16) unsigned short Bsm[2][2][512];

    const int tid   = threadIdx.x;
    const int lane  = tid & 31;
    const int half  = lane >> 4;
    const int widx  = tid >> 5;
    const int gwave = blockIdx.x * (NTHR / 32) + widx;
    int target = 0;

    // issuer role for async B staging (waves 0..3): (jI = local n-tile, pI = 16B part)
    const int jI = widx >> 1, pI = widx & 1;
    const int nt0 = blockIdx.x * 2;          // this WG owns n-tiles nt0, nt0+1
    const unsigned short* gB0 = Wcat
        + (size_t)((nt0 + jI) * 16 + (lane & 15)) * KX
        + (lane >> 4) * 16 + pI * 8;         // 16B per lane per chunk
    unsigned ldsB[2];
    ldsB[0] = lds_offset(&Bsm[0][jI][0]) + (unsigned)(lane * 32 + pI * 16);
    ldsB[1] = lds_offset(&Bsm[1][jI][0]) + (unsigned)(lane * 32 + pI * 16);

    for (int step = 0; step < LL; ++step) {
        // ---- S1: gates = xh @ Wcat^T.  WG = 2 n-tiles x 8 m-tiles; wave = 1 m-tile, 2 n-tiles.
        //      B fragments staged once per WG via global_load_async_to_lds_b128 (double buffered),
        //      shared by all 8 waves through ds_load; A fragment reused across both WMMAs. ----
        {
            const int mt = widx;
            if (widx < 4)
                async_ld_b128(ldsB[0], (unsigned long long)(uintptr_t)gB0);   // chunk 0
            v8f acc0 = {0.f, 0.f, 0.f, 0.f, 0.f, 0.f, 0.f, 0.f};
            v8f acc1 = {0.f, 0.f, 0.f, 0.f, 0.f, 0.f, 0.f, 0.f};
            for (int kc = 0; kc < KX / 32; ++kc) {
                if (widx < 4) {
                    if (kc + 1 < KX / 32) {
                        async_ld_b128(ldsB[(kc + 1) & 1],
                                      (unsigned long long)(uintptr_t)(gB0 + (kc + 1) * 32));
                        asm volatile("s_wait_asynccnt 0x1" ::: "memory");
                    } else {
                        asm volatile("s_wait_asynccnt 0x0" ::: "memory");
                    }
                }
                __syncthreads();                       // staged B visible to all waves
                v16bf a  = load_A(xh, mt * 16, kc * 32, KX);
                v16bf b0 = load_B_sm(&Bsm[kc & 1][0][0]);
                v16bf b1 = load_B_sm(&Bsm[kc & 1][1][0]);
                acc0 = wmma_bf16(a, b0, acc0);
                acc1 = wmma_bf16(a, b1, acc1);
                __syncthreads();                       // reads done before buffer reuse
            }
            store_C(gates, mt * 16, nt0 * 16, G4, acc0);
            store_C(gates, mt * 16, (nt0 + 1) * 16, G4, acc1);
        }
        gbar(bar, target);

        // ---- S2: LSTM cell elementwise (i,f,g,o order) ----
        for (int i = blockIdx.x * NTHR + tid; i < BB * HH; i += NWG * NTHR) {
            int b = i >> 9, j = i & 511;
            const float* gr = gates + (size_t)b * G4;
            float gi = gr[j]        + bias[j];
            float gf = gr[512 + j]  + bias[512 + j];
            float gg = gr[1024 + j] + bias[1024 + j];
            float go = gr[1536 + j] + bias[1536 + j];
            float cn = sigm(gf) * cst[i] + sigm(gi) * tanhf(gg);
            float hn = sigm(go) * tanhf(cn);
            cst[i] = cn;
            xh[(size_t)b * KX + 544 + j] = f2bf(hn);
        }
        gbar(bar, target);

        // ---- S3: q = relu(h @ phi_W^T + phi_b)  (128 tiles, K=512) ----
        for (int t = gwave; t < (BB / 16) * (MLPD / 16); t += NWAVES) {
            int mt = t >> 4, nt = t & 15;
            v8f acc = {0.f, 0.f, 0.f, 0.f, 0.f, 0.f, 0.f, 0.f};
#pragma unroll 4
            for (int k0 = 0; k0 < HH; k0 += 32) {
                v16bf a = load_A(xh, mt * 16, 544 + k0, KX);   // h region of xh
                v16bf b = load_B(phiW, nt * 16, k0, HH);
                acc = wmma_bf16(a, b, acc);
            }
            int col = nt * 16 + (lane & 15);
            float bv = phi_b[col];
#pragma unroll
            for (int v = 0; v < 8; ++v) {
                float r = acc[v] + bv;
                qbuf[(size_t)(mt * 16 + half * 8 + v) * MLPD + col] = r > 0.f ? r : 0.f;
            }
        }
        gbar(bar, target);

        // ---- S4: attention (energy -> softmax -> context) + next token, 2 batches per WG ----
        for (int bb = 0; bb < 2; ++bb) {
            int b = blockIdx.x * 2 + bb;
            sm_q[tid] = qbuf[(size_t)b * MLPD + tid];
            __syncthreads();
            // energy[t] = dot(q[b], cl[b,t,:])
            for (int t = tid; t < TT; t += NTHR) {
                const unsigned short* clr = cl + ((size_t)b * TT + t) * MLPD;
                float acc = 0.f;
#pragma unroll 8
                for (int m = 0; m < MLPD; m += 2) {
                    unsigned u = *(const unsigned*)(clr + m);
                    acc += sm_q[m]     * bf2f((unsigned short)(u & 0xffffu));
                    acc += sm_q[m + 1] * bf2f((unsigned short)(u >> 16));
                }
                sm_e[t] = acc;
            }
            __syncthreads();
            // softmax over T=400
            float lm = -1e30f;
            for (int t = tid; t < TT; t += NTHR) lm = fmaxf(lm, sm_e[t]);
            sm_red[tid] = lm; __syncthreads();
            for (int o = NTHR / 2; o > 0; o >>= 1) {
                if (tid < o) sm_red[tid] = fmaxf(sm_red[tid], sm_red[tid + o]);
                __syncthreads();
            }
            float mx = sm_red[0]; __syncthreads();
            float ls = 0.f;
            for (int t = tid; t < TT; t += NTHR) {
                float e = __expf(sm_e[t] - mx);
                sm_e[t] = e; ls += e;
            }
            sm_red[tid] = ls; __syncthreads();
            for (int o = NTHR / 2; o > 0; o >>= 1) {
                if (tid < o) sm_red[tid] += sm_red[tid + o];
                __syncthreads();
            }
            float inv = 1.0f / sm_red[0]; __syncthreads();
            for (int t = tid; t < TT; t += NTHR) sm_e[t] *= inv;
            __syncthreads();
            // context[d] = sum_t score[t] * lf[b,t,d]  -> xh context slot (bf16)
            for (int d = tid; d < HH; d += NTHR) {
                const unsigned short* lfr = lfb + (size_t)b * TT * HH + d;
                float acc = 0.f;
#pragma unroll 4
                for (int t = 0; t < TT; ++t) acc += sm_e[t] * bf2f(lfr[(size_t)t * HH]);
                xh[(size_t)b * KX + 32 + d] = f2bf(acc);
            }
            // next input token (teacher forcing)
            if (tid < NL) {
                int gv = gt[b * LL + step];
                xh[(size_t)b * KX + tid] = (tid == gv) ? f2bf(1.0f) : (unsigned short)0;
            }
            __syncthreads();
        }
        gbar(bar, target);

        // ---- S5: logits = [context|h] @ fcW^T + fc_b  (16 tiles, K=1024) ----
        if (gwave < (BB / 16) * (NL / 16)) {
            int mt = gwave >> 1, nt = gwave & 1;
            v8f acc = {0.f, 0.f, 0.f, 0.f, 0.f, 0.f, 0.f, 0.f};
#pragma unroll 4
            for (int k0 = 0; k0 < 1024; k0 += 32) {
                v16bf a = load_A(xh, mt * 16, 32 + k0, KX);    // skip tok region
                v16bf b = load_B(fcW, nt * 16, k0, 1024);
                acc = wmma_bf16(a, b, acc);
            }
            int col = nt * 16 + (lane & 15);
            float bv = fc_b[col];
#pragma unroll
            for (int v = 0; v < 8; ++v)
                logits[(size_t)(mt * 16 + half * 8 + v) * NL + col] = acc[v] + bv;
        }
        gbar(bar, target);

        // ---- S6: log_softmax over 32 labels; one wave per batch row ----
        if (gwave < BB) {
            int b = gwave;
            float x = logits[(size_t)b * NL + lane];
            float m = x;
            for (int o = 16; o > 0; o >>= 1) m = fmaxf(m, __shfl_xor(m, o, 32));
            float e = __expf(x - m);
            float s = e;
            for (int o = 16; o > 0; o >>= 1) s += __shfl_xor(s, o, 32);
            out[((size_t)b * LL + step) * NL + lane] = x - m - __logf(s);
        }
        gbar(bar, target);
    }
}

// ---------------- host launcher ----------------
extern "C" void kernel_launch(void* const* d_in, const int* in_sizes, int n_in,
                              void* d_out, int out_size, void* d_ws, size_t ws_size,
                              hipStream_t stream) {
    const float* lf    = (const float*)d_in[0];
    const int*   gt    = (const int*)d_in[1];
    const float* W_ih  = (const float*)d_in[2];
    const float* W_hh  = (const float*)d_in[3];
    const float* b_ih  = (const float*)d_in[4];
    const float* b_hh  = (const float*)d_in[5];
    const float* phi_W = (const float*)d_in[6];
    const float* phi_b = (const float*)d_in[7];
    const float* psi_W = (const float*)d_in[8];
    const float* psi_b = (const float*)d_in[9];
    const float* fc_W  = (const float*)d_in[10];
    const float* fc_b  = (const float*)d_in[11];
    float* out = (float*)d_out;

    // workspace carving (all offsets 256B-aligned)
    uint8_t* p = (uint8_t*)d_ws;
    auto take = [&](size_t bytes) {
        uint8_t* r = p;
        p += (bytes + 255) & ~(size_t)255;
        return r;
    };
    unsigned short* lfb   = (unsigned short*)take((size_t)BB * TT * HH * 2);   // 52.4 MB
    unsigned short* cl    = (unsigned short*)take((size_t)BB * TT * MLPD * 2); // 26.2 MB
    unsigned short* Wcat  = (unsigned short*)take((size_t)G4 * KX * 2);
    unsigned short* phiWb = (unsigned short*)take((size_t)MLPD * HH * 2);
    unsigned short* psiWb = (unsigned short*)take((size_t)MLPD * HH * 2);
    unsigned short* fcWb  = (unsigned short*)take((size_t)NL * 1024 * 2);
    float* bias   = (float*)take((size_t)G4 * 4);
    unsigned short* xh = (unsigned short*)take((size_t)BB * KX * 2);
    float* cst    = (float*)take((size_t)BB * HH * 4);
    float* gates  = (float*)take((size_t)BB * G4 * 4);
    float* qbuf   = (float*)take((size_t)BB * MLPD * 4);
    float* logits = (float*)take((size_t)BB * NL * 4);
    int*   bar    = (int*)take(256);

    hipMemsetAsync(bar, 0, 256, stream);

    prep_kernel<<<2048, 256, 0, stream>>>(W_ih, W_hh, b_ih, b_hh, phi_W, psi_W, fc_W, lf,
                                          Wcat, phiWb, psiWb, fcWb, bias, lfb);
    psi_gemm_kernel<<<512, 256, 0, stream>>>(lfb, psiWb, psi_b, cl);
    init_kernel<<<512, 256, 0, stream>>>(lfb, xh, cst);
    decode_kernel<<<NWG, NTHR, 0, stream>>>(lfb, cl, Wcat, phiWb, fcWb, bias, phi_b, fc_b,
                                            gt, xh, cst, gates, qbuf, logits, out, bar);
}
